// PhonemeDecoder_16595753632132
// MI455X (gfx1250) — matrix operand order repaired and checked by reading, not compiled, not run on hardware
//
#include <hip/hip_runtime.h>
#include <hip/hip_bf16.h>
#include <stdint.h>

// Problem constants (match reference)
#define VOCABN 128
#define HDIM   1024
#define LDIM   2
#define BDIM   512
#define TDIM   128
#define G3H    (3 * HDIM)   // 3072

typedef __bf16 bf16_t;
typedef bf16_t v16bf __attribute__((ext_vector_type(16)));
typedef float  v8f   __attribute__((ext_vector_type(8)));

union Frag32B { uint4 q[2]; v16bf v; };

// ---- CDNA5 async global->LDS path (guarded so a missing builtin can't break the build)
#if defined(__AMDGCN__) && __has_builtin(__builtin_amdgcn_global_load_async_to_lds_b128)
#define HAS_ASYNC_LDS 1
// Builtin signature: (global int4*, local int4*, imm offset, imm cpol)
typedef int v4i_vs __attribute__((vector_size(4 * sizeof(int))));
typedef __attribute__((address_space(1))) v4i_vs g_v4i;
typedef __attribute__((address_space(3))) v4i_vs l_v4i;
#else
#define HAS_ASYNC_LDS 0
#endif

#if HAS_ASYNC_LDS
#if __has_builtin(__builtin_amdgcn_s_wait_asynccnt)
#define WAIT_ASYNC(n) __builtin_amdgcn_s_wait_asynccnt(n)
#else
#define WAIT_ASYNC(n) asm volatile("s_wait_asynccnt %0" ::"n"(n))
#endif
#else
#define WAIT_ASYNC(n)
#endif

__device__ __forceinline__ void copy16_to_lds(const uint16_t* g, uint16_t* l) {
#if HAS_ASYNC_LDS
    __builtin_amdgcn_global_load_async_to_lds_b128((g_v4i*)g, (l_v4i*)l, 0, 0);
#else
    *(uint4*)l = *(const uint4*)g;
#endif
}

__device__ __forceinline__ uint16_t f32_to_bf16_rne(float f) {
    union { float f; uint32_t u; } x; x.f = f;
    uint32_t r = x.u + 0x7FFFu + ((x.u >> 16) & 1u);
    return (uint16_t)(r >> 16);
}

__device__ __forceinline__ float sigm(float x) { return 1.0f / (1.0f + __expf(-x)); }

// ---------------------------------------------------------------------------
// Setup kernels
// ---------------------------------------------------------------------------
__global__ void cvt_bf16_kernel(const float* __restrict__ src,
                                uint16_t* __restrict__ dst, int n) {
    int i = blockIdx.x * blockDim.x + threadIdx.x;
    if (i < n) dst[i] = f32_to_bf16_rne(src[i]);
}

__global__ void seed_h_kernel(const float* __restrict__ src,
                              float* __restrict__ hf,
                              uint16_t* __restrict__ hb, int n) {
    int i = blockIdx.x * blockDim.x + threadIdx.x;
    if (i < n) {
        float v = src[i];
        hf[i] = v;
        hb[i] = f32_to_bf16_rne(v);
    }
}

// tok[t*B + b]: t==0 -> inp[b], t>=1 -> target[b*T + t]
__global__ void build_tok_kernel(const int* __restrict__ inp,
                                 const int* __restrict__ target,
                                 int* __restrict__ tok) {
    int i = blockIdx.x * blockDim.x + threadIdx.x;
    if (i >= TDIM * BDIM) return;
    int t = i / BDIM, b = i % BDIM;
    tok[i] = (t == 0) ? inp[b] : target[b * TDIM + t];
}

// ---------------------------------------------------------------------------
// bf16 WMMA GEMM with LDS double-buffered staging (async global->LDS on CDNA5).
//   C(MxN,f32) = A(MxK,bf16 rm) * W(NxK,bf16 rm)^T + bias,  K = HDIM = 1024.
// Block = 256 thr = 8 waves arranged 4(M) x 2(N); wave tile 32x64 (2x4 WMMA).
// Block tile 128(M) x 128(N). Requires M,N multiples of 128.
// LDS rows padded to 80B so 64B-strided ds_load_b128 is bank-conflict-free.
// ---------------------------------------------------------------------------
#define LDS_ROW 80
#define KB      32
#define NKSTEP  (HDIM / KB)   // 32

__global__ __launch_bounds__(256)
void gemm_bf16_wmma(const uint16_t* __restrict__ A,
                    const uint16_t* __restrict__ W,
                    const float* __restrict__ bias,
                    float* __restrict__ C, int ldc) {
    __shared__ __align__(16) unsigned char smA[2][128 * LDS_ROW];
    __shared__ __align__(16) unsigned char smB[2][128 * LDS_ROW];

    const int tid  = threadIdx.x;
    const int lane = tid & 31;
    const int wave = tid >> 5;
    const int half = lane >> 4;   // 16-lane half of the wave
    const int lr   = lane & 15;
    const int wm   = wave >> 1;   // 0..3
    const int wn   = wave & 1;    // 0..1

    const int blockM = blockIdx.y * 128;
    const int blockN = blockIdx.x * 128;
    const uint16_t* Ablk = A + (size_t)blockM * HDIM;
    const uint16_t* Wblk = W + (size_t)blockN * HDIM;

    // Each thread stages 2 chunks of 16B per tile (512 chunks = 128 rows x 64B).
    const int row0 = tid >> 2, col0 = tid & 3;               // chunk tid
    const int row1 = (tid + 256) >> 2, col1 = (tid + 256) & 3;

    auto stage = [&](const uint16_t* gbase, unsigned char* lbuf, int kb) {
        copy16_to_lds(gbase + (size_t)row0 * HDIM + kb + col0 * 8,
                      (uint16_t*)(lbuf + row0 * LDS_ROW + col0 * 16));
        copy16_to_lds(gbase + (size_t)row1 * HDIM + kb + col1 * 8,
                      (uint16_t*)(lbuf + row1 * LDS_ROW + col1 * 16));
    };

    v8f acc[2][4];
#pragma unroll
    for (int i = 0; i < 2; ++i)
#pragma unroll
        for (int j = 0; j < 4; ++j) acc[i][j] = v8f{};

    // Prologue: stage k-step 0 into buffer 0.
    stage(Ablk, smA[0], 0);
    stage(Wblk, smB[0], 0);

    for (int k = 0; k < NKSTEP; ++k) {
        const int cur = k & 1;
        if (k + 1 < NKSTEP) {
            stage(Ablk, smA[cur ^ 1], (k + 1) * KB);
            stage(Wblk, smB[cur ^ 1], (k + 1) * KB);
            WAIT_ASYNC(4);   // the 4 async ops just issued may be in flight;
                             // everything for buffer `cur` has completed
        } else {
            WAIT_ASYNC(0);
        }
        __syncthreads();

        const unsigned char* bufA = &smA[cur][0];
        const unsigned char* bufB = &smB[cur][0];

        Frag32B a[2], b[4];
#pragma unroll
        for (int i = 0; i < 2; ++i) {
            int r = wm * 32 + i * 16 + lr;
            // A frag: elems 0-7 = K half*8.., elems 8-15 = K 16+half*8..
            a[i].q[0] = *(const uint4*)(bufA + r * LDS_ROW + half * 16);
            a[i].q[1] = *(const uint4*)(bufA + r * LDS_ROW + 32 + half * 16);
        }
#pragma unroll
        for (int j = 0; j < 4; ++j) {
            int n = wn * 64 + j * 16 + lr;
            // B frag: elems 0-7 = K half*8.., elems 8-15 = K 16+half*8..
            b[j].q[0] = *(const uint4*)(bufB + n * LDS_ROW + half * 16);
            b[j].q[1] = *(const uint4*)(bufB + n * LDS_ROW + 32 + half * 16);
        }
#pragma unroll
        for (int i = 0; i < 2; ++i)
#pragma unroll
            for (int j = 0; j < 4; ++j)
                acc[i][j] = __builtin_amdgcn_wmma_f32_16x16x32_bf16(
                    false, a[i].v, false, b[j].v, (short)0, acc[i][j], false, false);

        __syncthreads();   // protect buffer `cur` before it is restaged
    }

    // C/D layout: lanes 0-15 -> N=lr with M=vgpr; lanes 16-31 -> M=vgpr+8
#pragma unroll
    for (int i = 0; i < 2; ++i) {
#pragma unroll
        for (int j = 0; j < 4; ++j) {
            int col = blockN + wn * 64 + j * 16 + lr;
            float bv = bias ? bias[col] : 0.0f;
#pragma unroll
            for (int v = 0; v < 8; ++v) {
                int row = blockM + wm * 32 + i * 16 + half * 8 + v;
                C[(size_t)row * ldc + col] = acc[i][j][v] + bv;
            }
        }
    }
}

// ---------------------------------------------------------------------------
// GRU gate kernels (elementwise over B*H). gh/gx already include biases.
// ---------------------------------------------------------------------------
__global__ void gru_gate0_kernel(const int* __restrict__ tok_t,
                                 const float* __restrict__ Eproj,   // [V,3H], has b_ih[0]
                                 const float* __restrict__ gh,      // [B,3H], has b_hh[0]
                                 float* __restrict__ hf,
                                 uint16_t* __restrict__ hb) {
    int i = blockIdx.x * blockDim.x + threadIdx.x;
    if (i >= BDIM * HDIM) return;
    int b = i / HDIM, h = i % HDIM;
    const float* gx = Eproj + (size_t)tok_t[b] * G3H;
    const float* gr = gh + (size_t)b * G3H;
    float r = sigm(gx[h] + gr[h]);
    float z = sigm(gx[HDIM + h] + gr[HDIM + h]);
    float n = tanhf(gx[2 * HDIM + h] + r * gr[2 * HDIM + h]);
    float hn = (1.0f - z) * n + z * hf[i];
    hf[i] = hn;
    hb[i] = f32_to_bf16_rne(hn);
}

__global__ void gru_gate1_kernel(const float* __restrict__ gx,
                                 const float* __restrict__ gh,
                                 float* __restrict__ hf,
                                 uint16_t* __restrict__ hb) {
    int i = blockIdx.x * blockDim.x + threadIdx.x;
    if (i >= BDIM * HDIM) return;
    int b = i / HDIM, h = i % HDIM;
    const float* gxr = gx + (size_t)b * G3H;
    const float* ghr = gh + (size_t)b * G3H;
    float r = sigm(gxr[h] + ghr[h]);
    float z = sigm(gxr[HDIM + h] + ghr[HDIM + h]);
    float n = tanhf(gxr[2 * HDIM + h] + r * ghr[2 * HDIM + h]);
    float hn = (1.0f - z) * n + z * hf[i];
    hf[i] = hn;
    hb[i] = f32_to_bf16_rne(hn);
}

// ---------------------------------------------------------------------------
extern "C" void kernel_launch(void* const* d_in, const int* in_sizes, int n_in,
                              void* d_out, int out_size, void* d_ws, size_t ws_size,
                              hipStream_t stream) {
    const int*   inp    = (const int*)  d_in[0];   // (B,1)
    const int*   target = (const int*)  d_in[1];   // (B,T)
    const float* hidden = (const float*)d_in[2];   // (L,B,H)
    const float* emb    = (const float*)d_in[3];   // (V,H)
    const float* W_ih   = (const float*)d_in[4];   // (L,3H,H)
    const float* W_hh   = (const float*)d_in[5];   // (L,3H,H)
    const float* b_ih   = (const float*)d_in[6];   // (L,3H)
    const float* b_hh   = (const float*)d_in[7];   // (L,3H)
    float*       out    = (float*)d_out;           // (B,T,V)

    // ---- workspace layout ----
    char* ws = (char*)d_ws;
    const size_t nW  = (size_t)LDIM * G3H * HDIM;
    const size_t nWl = (size_t)G3H * HDIM;
    const size_t nE  = (size_t)VOCABN * HDIM;
    const size_t nEp = (size_t)VOCABN * G3H;
    const size_t nH  = (size_t)LDIM * BDIM * HDIM;
    const size_t nBH = (size_t)BDIM * HDIM;
    const size_t nG  = (size_t)BDIM * G3H;

    uint16_t* WihB  = (uint16_t*)ws;  ws += nW * 2;
    uint16_t* WhhB  = (uint16_t*)ws;  ws += nW * 2;
    uint16_t* EmbB  = (uint16_t*)ws;  ws += nE * 2;
    float*    Eproj = (float*)ws;     ws += nEp * 4;
    float*    hF    = (float*)ws;     ws += nH * 4;
    uint16_t* hB    = (uint16_t*)ws;  ws += nH * 2;
    float*    gh0   = (float*)ws;     ws += nG * 4;
    float*    gx1   = (float*)ws;     ws += nG * 4;
    float*    gh1   = (float*)ws;     ws += nG * 4;
    int*      tok   = (int*)ws;       ws += (size_t)TDIM * BDIM * 4;

    float*    h0F = hF;   float*    h1F = hF + nBH;
    uint16_t* h0B = hB;   uint16_t* h1B = hB + nBH;

    const int THREADS = 256;
    auto blocks = [](size_t n) { return (unsigned)((n + 255) / 256); };

    // ---- per-call setup: quantize weights, seed state, token table ----
    cvt_bf16_kernel<<<blocks(nW), THREADS, 0, stream>>>(W_ih, WihB, (int)nW);
    cvt_bf16_kernel<<<blocks(nW), THREADS, 0, stream>>>(W_hh, WhhB, (int)nW);
    cvt_bf16_kernel<<<blocks(nE), THREADS, 0, stream>>>(emb,  EmbB, (int)nE);
    seed_h_kernel  <<<blocks(nH), THREADS, 0, stream>>>(hidden, hF, hB, (int)nH);
    build_tok_kernel<<<blocks((size_t)TDIM * BDIM), THREADS, 0, stream>>>(inp, target, tok);

    // E_proj = emb @ W_ih[0]^T + b_ih[0]  (128 x 3072): replaces the big
    // per-token layer-0 input GEMM with a gather.
    gemm_bf16_wmma<<<dim3(G3H / 128, VOCABN / 128), THREADS, 0, stream>>>(
        EmbB, WihB, b_ih, Eproj, G3H);

    const dim3 gemmGrid(G3H / 128, BDIM / 128);    // 24 x 4
    const dim3 decGrid(VOCABN / 128, BDIM / 128);  // 1 x 4

    for (int t = 0; t < TDIM; ++t) {
        // layer 0 recurrent: gh0 = h0 @ W_hh[0]^T + b_hh[0]
        gemm_bf16_wmma<<<gemmGrid, THREADS, 0, stream>>>(h0B, WhhB, b_hh, gh0, G3H);
        gru_gate0_kernel<<<blocks(nBH), THREADS, 0, stream>>>(
            tok + (size_t)t * BDIM, Eproj, gh0, h0F, h0B);

        // layer 1 input + recurrent
        gemm_bf16_wmma<<<gemmGrid, THREADS, 0, stream>>>(h0B, WihB + nWl, b_ih + G3H, gx1, G3H);
        gemm_bf16_wmma<<<gemmGrid, THREADS, 0, stream>>>(h1B, WhhB + nWl, b_hh + G3H, gh1, G3H);
        gru_gate1_kernel<<<blocks(nBH), THREADS, 0, stream>>>(gx1, gh1, h1F, h1B);

        // tied-weight decode: logits[:, t, :] = h1 @ emb^T  (512 x 128)
        gemm_bf16_wmma<<<decGrid, THREADS, 0, stream>>>(
            h1B, EmbB, nullptr, out + (size_t)t * VOCABN, TDIM * VOCABN);
    }
}
